// GnnMP_69131793596533
// MI455X (gfx1250) — matrix operand order repaired and checked by reading, not compiled, not run on hardware
//
#include <hip/hip_runtime.h>
#include <hip/hip_bf16.h>

// ---- problem constants (from reference) ----
#define Bq      4
#define NPG     5000
#define Nn      20000          // B*NPG
#define IN_DIM  16
#define HID     128
#define Hh      8
#define DHd     16
#define DEG     32
#define NTYPE   12
#define NLAYERS 4
#define Ee      660000         // N*DEG + N
#define NFC     656            // 16 + 5*128
#define OUTC    1312           // 2*NFC

typedef __attribute__((ext_vector_type(16))) _Float16 v16h;
typedef __attribute__((ext_vector_type(8)))  float    v8f;

union V16H { v16h v; float4 q[2]; };

__device__ __forceinline__ float gelu_f(float x) {
  const float k0 = 0.7978845608028654f;   // sqrt(2/pi)
  float x3 = x * x * x;
  return 0.5f * x * (1.0f + tanhf(k0 * (x + 0.044715f * x3)));
}
__device__ __forceinline__ unsigned f2key(float f) {
  unsigned u = __float_as_uint(f);
  return (u & 0x80000000u) ? ~u : (u | 0x80000000u);
}
__device__ __forceinline__ float key2f(unsigned k) {
  return (k & 0x80000000u) ? __uint_as_float(k ^ 0x80000000u)
                           : __uint_as_float(~k);
}
__device__ __forceinline__ float lrelu02(float x) {
  return (x >= 0.f) ? x : 0.2f * x;
}

// ---- precompute per-edge-type attention bias: ea_type[t][h] ----
__global__ void ea_type_kernel(const float* __restrict__ edge_table,
                               const float* __restrict__ a_edge,
                               float* __restrict__ eat) {
  int idx = threadIdx.x;                    // 0..95
  if (idx >= NTYPE * Hh) return;
  int t = idx / Hh, h = idx % Hh;
  float s = 0.f;
  #pragma unroll
  for (int d = 0; d < DHd; ++d)
    s += edge_table[t * HID + h * DHd + d] * a_edge[h * DHd + d];
  eat[idx] = s;
}

// ---- pack W_conv (fp32 row-major KxN=128x128) into WMMA B-fragment order (f16) ----
__global__ void pack_w_kernel(const float* __restrict__ W, _Float16* __restrict__ wfrag) {
  int idx = blockIdx.x * blockDim.x + threadIdx.x;     // 16384
  if (idx >= HID * HID) return;
  int i    = idx & 15;
  int lane = (idx >> 4) & 31;
  int kb   = (idx >> 9) & 3;
  int ct   = idx >> 11;
  int K   = kb * 32 + ((lane < 16) ? 0 : 8) + (i & 7) + ((i >> 3) << 4);
  int col = ct * 16 + (lane & 15);
  wfrag[idx] = (_Float16)W[K * HID + col];
}

// ---- node embedder: masked feat -> out[:, :16]; h0 = gelu(feat@W_embed+b) ----
__global__ void embed_kernel(const float* __restrict__ feat,
                             const float* __restrict__ Wemb,
                             const float* __restrict__ bemb,
                             float* __restrict__ out,
                             _Float16* __restrict__ h16) {
  int idx = blockIdx.x * blockDim.x + threadIdx.x;     // N*128
  if (idx >= Nn * HID) return;
  int n = idx >> 7, c = idx & 127;
  // masked columns: 2,3 and 10..15 are zero -> only {0,1,4..9} contribute
  const int cols[8] = {0, 1, 4, 5, 6, 7, 8, 9};
  float s = bemb[c];
  #pragma unroll
  for (int k = 0; k < 8; ++k)
    s += feat[n * IN_DIM + cols[k]] * Wemb[cols[k] * HID + c];
  float h0 = gelu_f(s);
  size_t row = (size_t)n * OUTC;
  out[row + 16 + c] = h0;
  h16[idx] = (_Float16)h0;
  if (c < IN_DIM) {
    bool masked = (c >= 2 && c < 4) || (c >= 10);
    out[row + c] = masked ? 0.f : feat[n * IN_DIM + c];
  }
}

// ---- per-layer clear of agg / denom / mkey ----
__global__ void clear_kernel(float* __restrict__ agg, float* __restrict__ den,
                             unsigned* __restrict__ mkey) {
  int idx = blockIdx.x * blockDim.x + threadIdx.x;     // N*128
  if (idx >= Nn * HID) return;
  agg[idx] = 0.f;
  if (idx < Nn * Hh) { den[idx] = 0.f; mkey[idx] = 0u; }
}

// ---- z = h16 @ W_conv via v_wmma_f32_16x16x32_f16, one 16x16 tile per wave ----
__global__ void conv_wmma_kernel(const _Float16* __restrict__ h16,
                                 const _Float16* __restrict__ wfrag,
                                 float* __restrict__ z) {
  int wave = (blockIdx.x * blockDim.x + threadIdx.x) >> 5;  // 0..9999
  int lane = threadIdx.x & 31;
  int rowTile = wave >> 3;                                  // 0..1249
  int colTile = wave & 7;                                   // 0..7
  int row = rowTile * 16 + (lane & 15);
  int klo = (lane < 16) ? 0 : 8;
  v8f acc = {};
  #pragma unroll
  for (int kb = 0; kb < 4; ++kb) {
    // A fragment (16x32 f16): lane holds K = kb*32 + {klo..klo+7, 16+klo..16+klo+7}
    const _Float16* ap = h16 + row * HID + kb * 32 + klo;
    V16H ua, ub;
    ua.q[0] = *reinterpret_cast<const float4*>(ap);
    ua.q[1] = *reinterpret_cast<const float4*>(ap + 16);
    // B fragment: pre-swizzled, 16 contiguous f16 per lane
    const _Float16* bp = wfrag + ((colTile * 4 + kb) * 32 + lane) * 16;
    ub.q[0] = *reinterpret_cast<const float4*>(bp);
    ub.q[1] = *reinterpret_cast<const float4*>(bp + 8);
    acc = __builtin_amdgcn_wmma_f32_16x16x32_f16(
        false, ua.v, false, ub.v, (short)0, acc, false, false);
  }
  // C/D layout: N = lane%16; M = r + 8*(lane/16)
  int n = colTile * 16 + (lane & 15);
  int mbase = rowTile * 16 + ((lane < 16) ? 0 : 8);
  #pragma unroll
  for (int r = 0; r < 8; ++r)
    z[(size_t)(mbase + r) * HID + n] = acc[r];
}

// ---- per-node attention logit halves (vectorized z reads) ----
__global__ void node_logit_kernel(const float* __restrict__ z,
                                  const float* __restrict__ a_src,
                                  const float* __restrict__ a_dst,
                                  float* __restrict__ slog,
                                  float* __restrict__ dlog) {
  int idx = blockIdx.x * blockDim.x + threadIdx.x;     // N*8
  if (idx >= Nn * Hh) return;
  int n = idx >> 3, h = idx & 7;
  const float4* zp = reinterpret_cast<const float4*>(z + (size_t)n * HID + h * DHd);
  const float4* sp = reinterpret_cast<const float4*>(a_src + h * DHd);
  const float4* dp = reinterpret_cast<const float4*>(a_dst + h * DHd);
  float s = 0.f, d = 0.f;
  #pragma unroll
  for (int q = 0; q < 4; ++q) {
    float4 zv = zp[q], sv = sp[q], dv = dp[q];
    s += zv.x * sv.x + zv.y * sv.y + zv.z * sv.z + zv.w * sv.w;
    d += zv.x * dv.x + zv.y * dv.y + zv.z * dv.z + zv.w * dv.w;
  }
  slog[idx] = s; dlog[idx] = d;
}

// ---- edge logits + segment max: one thread per edge, all 8 heads ----
__global__ void edge_logit_kernel(const int* __restrict__ ei,
                                  const int* __restrict__ etype,
                                  const float* __restrict__ slog,
                                  const float* __restrict__ dlog,
                                  const float* __restrict__ eat,
                                  float* __restrict__ lg,
                                  unsigned* __restrict__ mkey) {
  int e = blockIdx.x * blockDim.x + threadIdx.x;       // E
  if (e >= Ee) return;
  int src = ei[e], dst = ei[Ee + e], t = etype[e];
  const float4* sp = reinterpret_cast<const float4*>(slog + src * Hh);
  const float4* dp = reinterpret_cast<const float4*>(dlog + dst * Hh);
  const float4* ep = reinterpret_cast<const float4*>(eat + t * Hh);
  float4 s0 = sp[0], s1 = sp[1], d0 = dp[0], d1 = dp[1], e0 = ep[0], e1 = ep[1];
  float4 x0, x1;
  x0.x = lrelu02(s0.x + d0.x + e0.x);
  x0.y = lrelu02(s0.y + d0.y + e0.y);
  x0.z = lrelu02(s0.z + d0.z + e0.z);
  x0.w = lrelu02(s0.w + d0.w + e0.w);
  x1.x = lrelu02(s1.x + d1.x + e1.x);
  x1.y = lrelu02(s1.y + d1.y + e1.y);
  x1.z = lrelu02(s1.z + d1.z + e1.z);
  x1.w = lrelu02(s1.w + d1.w + e1.w);
  float4* op = reinterpret_cast<float4*>(lg + (size_t)e * Hh);
  op[0] = x0; op[1] = x1;
  unsigned* mk = mkey + dst * Hh;
  atomicMax(mk + 0, f2key(x0.x));
  atomicMax(mk + 1, f2key(x0.y));
  atomicMax(mk + 2, f2key(x0.z));
  atomicMax(mk + 3, f2key(x0.w));
  atomicMax(mk + 4, f2key(x1.x));
  atomicMax(mk + 5, f2key(x1.y));
  atomicMax(mk + 6, f2key(x1.z));
  atomicMax(mk + 7, f2key(x1.w));
}

// ---- exp(logit - max) + segment sum: one thread per edge ----
__global__ void edge_exp_kernel(const int* __restrict__ ei,
                                float* __restrict__ lg,
                                const unsigned* __restrict__ mkey,
                                float* __restrict__ den) {
  int e = blockIdx.x * blockDim.x + threadIdx.x;       // E
  if (e >= Ee) return;
  int dst = ei[Ee + e];
  float4* lp = reinterpret_cast<float4*>(lg + (size_t)e * Hh);
  const uint4* mp = reinterpret_cast<const uint4*>(mkey + dst * Hh);
  float4 x0 = lp[0], x1 = lp[1];
  uint4 m0 = mp[0], m1 = mp[1];
  x0.x = expf(x0.x - key2f(m0.x));
  x0.y = expf(x0.y - key2f(m0.y));
  x0.z = expf(x0.z - key2f(m0.z));
  x0.w = expf(x0.w - key2f(m0.w));
  x1.x = expf(x1.x - key2f(m1.x));
  x1.y = expf(x1.y - key2f(m1.y));
  x1.z = expf(x1.z - key2f(m1.z));
  x1.w = expf(x1.w - key2f(m1.w));
  lp[0] = x0; lp[1] = x1;
  float* dn = den + dst * Hh;
  atomicAdd(dn + 0, x0.x);
  atomicAdd(dn + 1, x0.y);
  atomicAdd(dn + 2, x0.z);
  atomicAdd(dn + 3, x0.w);
  atomicAdd(dn + 4, x1.x);
  atomicAdd(dn + 5, x1.y);
  atomicAdd(dn + 6, x1.z);
  atomicAdd(dn + 7, x1.w);
}

// ---- agg[dst] += alpha * z[src]: one thread per (edge, 4-channel quad) ----
__global__ void aggregate_kernel(const int* __restrict__ ei,
                                 const float* __restrict__ lg,
                                 const float* __restrict__ den,
                                 const float* __restrict__ z,
                                 float* __restrict__ agg) {
  size_t idx = (size_t)blockIdx.x * blockDim.x + threadIdx.x;   // E*32
  if (idx >= (size_t)Ee * 32) return;
  int e = (int)(idx >> 5), q = (int)(idx & 31);
  int c0 = q << 2;                // channel quad start (stays inside one head)
  int h = q >> 2;
  int src = ei[e], dst = ei[Ee + e];
  float alpha = lg[(size_t)e * Hh + h] / (den[dst * Hh + h] + 1e-9f);
  float4 zv = *reinterpret_cast<const float4*>(z + (size_t)src * HID + c0);
  float* ap = agg + (size_t)dst * HID + c0;
  atomicAdd(ap + 0, alpha * zv.x);
  atomicAdd(ap + 1, alpha * zv.y);
  atomicAdd(ap + 2, alpha * zv.z);
  atomicAdd(ap + 3, alpha * zv.w);
}

// ---- residual update h <- h + gelu(agg); write next layer block + f16 copy ----
__global__ void update_kernel(const float* __restrict__ agg,
                              float* __restrict__ out,
                              _Float16* __restrict__ h16,
                              int layer) {
  int idx = blockIdx.x * blockDim.x + threadIdx.x;     // N*128
  if (idx >= Nn * HID) return;
  int n = idx >> 7, c = idx & 127;
  size_t row = (size_t)n * OUTC;
  float hp = out[row + 16 + layer * HID + c];
  float hn = hp + gelu_f(agg[idx]);
  out[row + 16 + (layer + 1) * HID + c] = hn;
  h16[idx] = (_Float16)hn;
}

// ---- graph max-pool: one block per (b, c) ----
__global__ void pool_kernel(const float* __restrict__ out, float* __restrict__ g) {
  __shared__ float sm[256];
  int bc = blockIdx.x;                                 // B*656
  int b = bc / NFC, c = bc % NFC;
  float m = -3.402823466e+38f;
  for (int p = threadIdx.x; p < NPG; p += blockDim.x) {
    float v = out[((size_t)b * NPG + p) * OUTC + c];
    m = fmaxf(m, v);
  }
  sm[threadIdx.x] = m;
  __syncthreads();
  for (int s = 128; s > 0; s >>= 1) {
    if (threadIdx.x < s) sm[threadIdx.x] = fmaxf(sm[threadIdx.x], sm[threadIdx.x + s]);
    __syncthreads();
  }
  if (threadIdx.x == 0) g[bc] = sm[0];
}

// ---- broadcast g_emb into out[:, :, 656:1312] ----
__global__ void broadcast_kernel(const float* __restrict__ g, float* __restrict__ out) {
  size_t idx = (size_t)blockIdx.x * blockDim.x + threadIdx.x;   // B*NPG*656
  if (idx >= (size_t)Nn * NFC) return;
  int n = (int)(idx / NFC), c = (int)(idx % NFC);
  int b = n / NPG;
  out[(size_t)n * OUTC + NFC + c] = g[b * NFC + c];
}

static inline size_t al256(size_t x) { return (x + 255) & ~(size_t)255; }

extern "C" void kernel_launch(void* const* d_in, const int* in_sizes, int n_in,
                              void* d_out, int out_size, void* d_ws, size_t ws_size,
                              hipStream_t stream) {
  const float* feat    = (const float*)d_in[0];
  const float* etab    = (const float*)d_in[1];
  const float* Wemb    = (const float*)d_in[2];
  const float* bemb    = (const float*)d_in[3];
  const float* Wconv   = (const float*)d_in[4];
  const float* a_src   = (const float*)d_in[5];
  const float* a_dst   = (const float*)d_in[6];
  const float* a_edge  = (const float*)d_in[7];
  const int*   ei      = (const int*)d_in[8];
  const int*   etype   = (const int*)d_in[9];
  float* out = (float*)d_out;

  // workspace carve-up (~49 MB total, fully L2-resident on MI455X)
  char* ws = (char*)d_ws;
  size_t off = 0;
  _Float16* h16   = (_Float16*)(ws + off); off += al256((size_t)Nn * HID * 2);
  _Float16* wfrag = (_Float16*)(ws + off); off += al256((size_t)HID * HID * 2);
  float*    z     = (float*)(ws + off);    off += al256((size_t)Nn * HID * 4);
  float*    slog  = (float*)(ws + off);    off += al256((size_t)Nn * Hh * 4);
  float*    dlog  = (float*)(ws + off);    off += al256((size_t)Nn * Hh * 4);
  float*    eat   = (float*)(ws + off);    off += al256((size_t)NTYPE * Hh * 4);
  float*    lg    = (float*)(ws + off);    off += al256((size_t)Ee * Hh * 4);
  unsigned* mkey  = (unsigned*)(ws + off); off += al256((size_t)Nn * Hh * 4);
  float*    den   = (float*)(ws + off);    off += al256((size_t)Nn * Hh * 4);
  float*    agg   = (float*)(ws + off);    off += al256((size_t)Nn * HID * 4);
  float*    g     = (float*)(ws + off);    off += al256((size_t)Bq * NFC * 4);
  (void)ws_size; (void)in_sizes; (void)n_in; (void)out_size;

  ea_type_kernel<<<1, 96, 0, stream>>>(etab, a_edge, eat);
  pack_w_kernel<<<(HID * HID + 255) / 256, 256, 0, stream>>>(Wconv, wfrag);
  embed_kernel<<<(Nn * HID + 255) / 256, 256, 0, stream>>>(feat, Wemb, bemb, out, h16);

  for (int l = 0; l < NLAYERS; ++l) {
    clear_kernel<<<(Nn * HID + 255) / 256, 256, 0, stream>>>(agg, den, mkey);
    conv_wmma_kernel<<<(Nn / 16) * (HID / 16) / 4, 128, 0, stream>>>(h16, wfrag, z);
    node_logit_kernel<<<(Nn * Hh + 255) / 256, 256, 0, stream>>>(z, a_src, a_dst, slog, dlog);
    edge_logit_kernel<<<(Ee + 255) / 256, 256, 0, stream>>>(ei, etype, slog, dlog, eat, lg, mkey);
    edge_exp_kernel<<<(Ee + 255) / 256, 256, 0, stream>>>(ei, lg, mkey, den);
    aggregate_kernel<<<(unsigned)(((size_t)Ee * 32 + 255) / 256), 256, 0, stream>>>(ei, lg, den, z, agg);
    update_kernel<<<(Nn * HID + 255) / 256, 256, 0, stream>>>(agg, out, h16, l);
  }

  pool_kernel<<<Bq * NFC, 256, 0, stream>>>(out, g);
  broadcast_kernel<<<(unsigned)(((size_t)Nn * NFC + 255) / 256), 256, 0, stream>>>(g, out);
}
